// EuclideanCodebook_58531814310487
// MI455X (gfx1250) — matrix operand order repaired and checked by reading, not compiled, not run on hardware
//
#include <hip/hip_runtime.h>

// ---------------------------------------------------------------------------
// EuclideanCodebook eval forward for MI455X (gfx1250, wave32, WMMA).
//   dist = -sqrt(clip(|x|^2 + |e|^2 - 2 x.e, 0))   [N=32768, C=8192], d=512
//   ind  = argmax_C dist ; quantize = embed[ind]
// GEMM done as bf16x3 split-precision WMMA (f32 accumulate).
// Pipeline: A panel resident in LDS; B streamed via async global->LDS with
// ping-pong buffers so the copy hides behind the WMMA work.
// ---------------------------------------------------------------------------

typedef __attribute__((ext_vector_type(16))) __bf16 v16bf;
typedef __attribute__((ext_vector_type(8)))  float  v8f;
typedef __attribute__((ext_vector_type(4)))  __bf16 v4bf;
typedef __attribute__((ext_vector_type(4)))  int    v4i;

#define DIMD   512
#define CODES  8192
#define NROWS  32768
#define BM     128
#define BN     128
#define KC     32
#define NKCH   (DIMD / KC)              // 16 K-chunks

// ---- LDS layout (bytes) for the main kernel -------------------------------
#define LDS_AH    0                     // [BM][DIMD] bf16 hi   = 131072
#define LDS_AL    131072                // [BM][DIMD] bf16 lo   = 131072
#define LDS_B0H   262144                // [BN][KC] bf16 hi, buf0 = 8192
#define LDS_B0L   270336                // [BN][KC] bf16 lo, buf0 = 8192
#define LDS_B1H   278528                // buf1 hi = 8192
#define LDS_B1L   286720                // buf1 lo = 8192
#define LDS_E2    294912                // [BN] f32 = 512
#define LDS_X2    295424                // [BM] f32 = 512
#define LDS_TOTAL 295936
// reduction buffers alias the A-panel (only used after all WMMA work):
#define LDS_REDD 0                      // [BM][32] f32 = 16384
#define LDS_REDI 16384                  // [BM][32] i32 = 16384

// gfx1250 async global->LDS copy path (existence proven by round-2 probe)
#if __has_builtin(__builtin_amdgcn_global_load_async_to_lds_b128) && \
    __has_builtin(__builtin_amdgcn_s_wait_asynccnt)
#define HAVE_ASYNC_LDS 1
#define AS1P(p) ((__attribute__((address_space(1))) v4i*)(uintptr_t)(p))
#define AS3P(p) ((__attribute__((address_space(3))) v4i*)(uintptr_t)(p))
#else
#define HAVE_ASYNC_LDS 0
#endif

#if __has_builtin(__builtin_amdgcn_sqrtf)
#define FAST_SQRT(x) __builtin_amdgcn_sqrtf(x)   // raw v_sqrt_f32, no IEEE fixup
#else
#define FAST_SQRT(x) __builtin_sqrtf(x)
#endif

union Frag { v16bf v; uint4 q[2]; };

__device__ __forceinline__ v8f wmma_bf16(const Frag& a, const Frag& b, v8f c) {
  return __builtin_amdgcn_wmma_f32_16x16x32_bf16(
      /*neg_a=*/false, a.v, /*neg_b=*/false, b.v,
      /*c_mod=*/(short)0, c, /*reuse_a=*/false, /*reuse_b=*/false);
}

// ---------------------------------------------------------------------------
// Kernel 1: per-row bf16 hi/lo split + sum-of-squares.  One wave per row.
// ---------------------------------------------------------------------------
__global__ void vq_prep_rows(const float* __restrict__ src,
                             __bf16* __restrict__ hi, __bf16* __restrict__ lo,
                             float* __restrict__ sq, int nrows) {
  const int wid  = (blockIdx.x * blockDim.x + threadIdx.x) >> 5;
  const int lane = threadIdx.x & 31;
  if (wid >= nrows) return;
  const float4* row = (const float4*)(src + (size_t)wid * DIMD);
  float ss = 0.0f;
#pragma unroll
  for (int i = 0; i < 4; ++i) {
    const int q = i * 32 + lane;                 // float4 index 0..127
    float4 v = row[q];
    ss += v.x * v.x + v.y * v.y + v.z * v.z + v.w * v.w;
    v4bf h, l;
    h.x = (__bf16)v.x; l.x = (__bf16)(v.x - (float)h.x);
    h.y = (__bf16)v.y; l.y = (__bf16)(v.y - (float)h.y);
    h.z = (__bf16)v.z; l.z = (__bf16)(v.z - (float)h.z);
    h.w = (__bf16)v.w; l.w = (__bf16)(v.w - (float)h.w);
    *(v4bf*)(hi + (size_t)wid * DIMD + q * 4) = h;
    *(v4bf*)(lo + (size_t)wid * DIMD + q * 4) = l;
  }
#pragma unroll
  for (int m = 16; m; m >>= 1) ss += __shfl_xor(ss, m, 32);
  if (lane == 0) sq[wid] = ss;
}

// ---------------------------------------------------------------------------
// Kernel 2: per 128-row band: full-C WMMA GEMM + dist write + fused argmax.
// 256 threads = 8 waves; waves tile 128x128 as 4(M) x 2(N), 32x64 each.
// One workgroup per WGP (296 KB LDS) -> launch_bounds(256,1) for full VGPRs.
// ---------------------------------------------------------------------------
__global__ void __launch_bounds__(256, 1)
vq_dist_argmax(const __bf16* __restrict__ xh, const __bf16* __restrict__ xl,
               const __bf16* __restrict__ eh, const __bf16* __restrict__ el,
               const float* __restrict__ x2, const float* __restrict__ e2,
               float* __restrict__ dist_out, float* __restrict__ ind_out,
               int* __restrict__ idx_ws) {
  extern __shared__ char smem[];
  __bf16* As_h = (__bf16*)(smem + LDS_AH);
  __bf16* As_l = (__bf16*)(smem + LDS_AL);
  __bf16* Bh[2] = { (__bf16*)(smem + LDS_B0H), (__bf16*)(smem + LDS_B1H) };
  __bf16* Bl[2] = { (__bf16*)(smem + LDS_B0L), (__bf16*)(smem + LDS_B1L) };
  float*  e2s  = (float*)(smem + LDS_E2);
  float*  x2s  = (float*)(smem + LDS_X2);
  float*  redD = (float*)(smem + LDS_REDD);
  int*    redI = (int*)(smem + LDS_REDI);

  const int t      = threadIdx.x;
  const int lane   = t & 31;
  const int wid    = t >> 5;
  const int wave_m = wid & 3;        // 0..3 -> 32-row slice
  const int wave_n = wid >> 2;       // 0..1 -> 64-col slice
  const int half   = lane >> 4;      // lane group (ISA A/B fragment layout)
  const int l16    = lane & 15;
  const int m0     = blockIdx.x * BM;

  // fill one 128x32 embed tile (hi+lo) into buffer `buf` for K-chunk k0
  auto issueB = [&](int c0, int k0, int buf) {
#pragma unroll
    for (int q = t; q < BN * KC / 8; q += 256) {   // 16B chunks, 2/thread/half
      const int code = q >> 2, part = q & 3;
      const size_t gofs = (size_t)(c0 + code) * DIMD + k0 + part * 8;
      const int    sofs = code * KC + part * 8;
#if HAVE_ASYNC_LDS
      __builtin_amdgcn_global_load_async_to_lds_b128(
          AS1P(eh + gofs), AS3P(Bh[buf] + sofs), 0, 0);
      __builtin_amdgcn_global_load_async_to_lds_b128(
          AS1P(el + gofs), AS3P(Bl[buf] + sofs), 0, 0);
#else
      *(uint4*)(Bh[buf] + sofs) = *(const uint4*)(eh + gofs);
      *(uint4*)(Bl[buf] + sofs) = *(const uint4*)(el + gofs);
      __builtin_prefetch(eh + gofs + KC, 0, 0);
      __builtin_prefetch(el + gofs + KC, 0, 0);
#endif
    }
  };

  // ---- park the 128x512 hi/lo x-panel in LDS (256 KB, once per band) ----
  {
    const uint4* gh = (const uint4*)(xh + (size_t)m0 * DIMD);
    const uint4* gl = (const uint4*)(xl + (size_t)m0 * DIMD);
    uint4* sh = (uint4*)As_h;
    uint4* sl = (uint4*)As_l;
    for (int i = t; i < BM * DIMD * 2 / 16; i += 256) { sh[i] = gh[i]; sl[i] = gl[i]; }
  }
  if (t < BM) x2s[t] = x2[m0 + t];
  __syncthreads();

  float bestD[16];
  int   bestI[16];
#pragma unroll
  for (int e = 0; e < 16; ++e) { bestD[e] = -3.4e38f; bestI[e] = 0; }

  v8f acc[2][4];

  for (int ct = 0; ct < CODES / BN; ++ct) {
    const int c0 = ct * BN;
    __syncthreads();                       // prev tile done reading e2s/B bufs
    if (t < BN) e2s[t] = e2[c0 + t];
    issueB(c0, 0, 0);                      // prologue fill of buf0
#pragma unroll
    for (int mt = 0; mt < 2; ++mt)
#pragma unroll
      for (int nt = 0; nt < 4; ++nt)
        acc[mt][nt] = (v8f){0.f, 0.f, 0.f, 0.f, 0.f, 0.f, 0.f, 0.f};

    for (int kk = 0; kk < NKCH; ++kk) {
      const int k0  = kk * KC;
      const int buf = kk & 1;
#if HAVE_ASYNC_LDS
      __builtin_amdgcn_s_wait_asynccnt(0); // my chunk of buf[kk] has landed
#endif
      __syncthreads();                     // everyone's chunk landed; prev
                                           // compute done reading buf^1
      if (kk + 1 < NKCH) issueB(c0, k0 + KC, buf ^ 1);  // overlap next copy

      // A fragments: 16-bit 16x32 layout -> lane<16: K 0..7 & 16..23 of row l16
      Frag ah[2], al[2];
#pragma unroll
      for (int mt = 0; mt < 2; ++mt) {
        const int mrow = wave_m * 32 + mt * 16 + l16;
        const __bf16* pa = As_h + (size_t)mrow * DIMD + k0 + half * 8;
        ah[mt].q[0] = *(const uint4*)pa;
        ah[mt].q[1] = *(const uint4*)(pa + 16);
        const __bf16* pl = As_l + (size_t)mrow * DIMD + k0 + half * 8;
        al[mt].q[0] = *(const uint4*)pl;
        al[mt].q[1] = *(const uint4*)(pl + 16);
      }
      // One N-tile of B live at a time (keeps VGPR pressure low, no spills).
#pragma unroll
      for (int nt = 0; nt < 4; ++nt) {
        Frag bh, bl;
        const int col = wave_n * 64 + nt * 16 + l16;
        const __bf16* pb = Bh[buf] + col * KC + half * 8;
        bh.q[0] = *(const uint4*)pb;
        bh.q[1] = *(const uint4*)(pb + 16);
        const __bf16* pl = Bl[buf] + col * KC + half * 8;
        bl.q[0] = *(const uint4*)pl;
        bl.q[1] = *(const uint4*)(pl + 16);
        // bf16x3: xe ~= lo*hi + hi*lo + hi*hi, accumulated in f32
#pragma unroll
        for (int mt = 0; mt < 2; ++mt) {
          v8f c = acc[mt][nt];
          c = wmma_bf16(al[mt], bh, c);
          c = wmma_bf16(ah[mt], bl, c);
          c = wmma_bf16(ah[mt], bh, c);
          acc[mt][nt] = c;
        }
      }
    }

    // ---- postprocess: dist = -sqrt(clip(x2+e2-2xe,0)); NT store; argmax ----
#pragma unroll
    for (int mt = 0; mt < 2; ++mt)
#pragma unroll
      for (int nt = 0; nt < 4; ++nt) {
        const int colT = wave_n * 64 + nt * 16 + l16;
        const float e2v = e2s[colT];
        const int gcol = c0 + colT;
#pragma unroll
        for (int j = 0; j < 8; ++j) {
          const int mloc = wave_m * 32 + mt * 16 + half * 8 + j;
          float d2 = x2s[mloc] + e2v - 2.0f * acc[mt][nt][j];
          d2 = d2 < 0.0f ? 0.0f : d2;
          const float dv = -FAST_SQRT(d2);
          __builtin_nontemporal_store(dv, dist_out + (size_t)(m0 + mloc) * CODES + gcol);
          const int e = mt * 8 + j;
          if (dv > bestD[e]) { bestD[e] = dv; bestI[e] = gcol; }
        }
      }
  }

  // ---- band-local argmax reduction (A panel no longer needed: alias LDS) ----
  __syncthreads();
#pragma unroll
  for (int mt = 0; mt < 2; ++mt)
#pragma unroll
    for (int j = 0; j < 8; ++j) {
      const int e = mt * 8 + j;
      const int mloc = wave_m * 32 + mt * 16 + half * 8 + j;
      const int slot = wave_n * 16 + l16;
      redD[mloc * 32 + slot] = bestD[e];
      redI[mloc * 32 + slot] = bestI[e];
    }
  __syncthreads();
  if (t < BM) {
    float bd = redD[t * 32];
    int   bi = redI[t * 32];
    for (int s = 1; s < 32; ++s) {
      const float dv = redD[t * 32 + s];
      const int   iv = redI[t * 32 + s];
      if (dv > bd || (dv == bd && iv < bi)) { bd = dv; bi = iv; }  // ties -> lowest idx
    }
    ind_out[m0 + t] = (float)bi;
    idx_ws[m0 + t]  = bi;
  }
}

// ---------------------------------------------------------------------------
// Kernel 3: quantize = embed[idx] gather (f32, one wave per row).
// ---------------------------------------------------------------------------
__global__ void vq_gather(const float* __restrict__ embed, const int* __restrict__ idx,
                          float* __restrict__ out) {
  const int wid  = (blockIdx.x * blockDim.x + threadIdx.x) >> 5;
  const int lane = threadIdx.x & 31;
  const int code = idx[wid];
  const float4* s = (const float4*)(embed + (size_t)code * DIMD);
  float4*       d = (float4*)(out + (size_t)wid * DIMD);
#pragma unroll
  for (int i = 0; i < 4; ++i) d[i * 32 + lane] = s[i * 32 + lane];
}

// ---------------------------------------------------------------------------
extern "C" void kernel_launch(void* const* d_in, const int* in_sizes, int n_in,
                              void* d_out, int out_size, void* d_ws, size_t ws_size,
                              hipStream_t stream) {
  const float* x     = (const float*)d_in[0];   // [8,4096,512]
  const float* embed = (const float*)d_in[1];   // [1,8192,512]

  char* ws = (char*)d_ws;                       // ~84 MB of scratch used
  __bf16* xh  = (__bf16*)(ws + 0);
  __bf16* xl  = (__bf16*)(ws + 33554432);
  __bf16* eh  = (__bf16*)(ws + 67108864);
  __bf16* el  = (__bf16*)(ws + 75497472);
  float*  x2  = (float*)(ws + 83886080);
  float*  e2  = (float*)(ws + 84017152);
  int*    idw = (int*)(ws + 84049920);

  float* out      = (float*)d_out;
  float* q_out    = out;                         // 16,777,216
  float* ind_out  = out + 16777216;              // 32,768
  float* dist_out = out + 16809984;              // 268,435,456

  vq_prep_rows<<<NROWS / 8, 256, 0, stream>>>(x, xh, xl, x2, NROWS);
  vq_prep_rows<<<CODES / 8, 256, 0, stream>>>(embed, eh, el, e2, CODES);
  vq_dist_argmax<<<NROWS / BM, 256, LDS_TOTAL, stream>>>(xh, xl, eh, el, x2, e2,
                                                         dist_out, ind_out, idw);
  vq_gather<<<NROWS / 8, 256, 0, stream>>>(embed, idw, q_out);
}
